// MultiHeadAttention_12670153523418
// MI455X (gfx1250) — compile-verified, matrix operand order checked by reading, not run on hardware
//
#include <hip/hip_runtime.h>
#include <cstdint>

// ---------------------------------------------------------------------------
// MultiHeadAttention forward for MI455X (gfx1250), wave32 + WMMA f16->f32,
// async global->LDS staging (ASYNCcnt), double-buffered LDS in main GEMMs,
// fused partial-softmax in scores + fused normalize in context (saves a full
// 512MB softmax pass over the 256MB attn tensor; HBM-bound at 23.3 TB/s).
// d_out = [out (4096*1024 f32)] ++ [attn (4*16*1024*1024 f32)]
// ---------------------------------------------------------------------------

typedef __attribute__((ext_vector_type(16))) _Float16 v16h;
typedef __attribute__((ext_vector_type(8)))  _Float16 v8h;
typedef __attribute__((ext_vector_type(8)))  float    v8f;
typedef __attribute__((ext_vector_type(4)))  float    v4f;

#define S_LEN  1024
#define DMODEL 1024
#define NHEAD  16
#define DK     64
#define BATCH  4
#define MROWS  (BATCH * S_LEN)   // 4096
#define NKT    (S_LEN / 64)      // 16 k-tiles per row

// ---- WMMA wrapper ---------------------------------------------------------
__device__ __forceinline__ v8f wmma16(v16h a, v16h b, v8f c) {
  return __builtin_amdgcn_wmma_f32_16x16x32_f16(
      /*neg_a=*/false, a, /*neg_b=*/false, b,
      /*c_mod=*/(short)0, c, /*reuse_a=*/false, /*reuse_b=*/false);
}

// ---- CDNA5 async global->LDS copy (ASYNCcnt) ------------------------------
__device__ __forceinline__ uint32_t lds_addr_of(const void* p) {
  return (uint32_t)(uintptr_t)p;   // ADDR[31:0] of LDS generic addr = offset
}
__device__ __forceinline__ void async_b128(const void* g, const void* lds) {
  asm volatile("global_load_async_to_lds_b128 %0, %1, off"
               :: "v"(lds_addr_of(lds)), "v"(g)
               : "memory");
}
__device__ __forceinline__ void async_wait0() {
#if __has_builtin(__builtin_amdgcn_s_wait_asynccnt)
  __builtin_amdgcn_s_wait_asynccnt(0);
#else
  asm volatile("s_wait_asynccnt 0" ::: "memory");
#endif
}

// ---- Fragment load, row-major source (LDS or global) ----------------------
// 16-bit A 16x32 layout (ISA 7.12.2): lane L holds row L%16;
// K: [ (L/16)*8 .. +8 ) and [ 16 + (L/16)*8 .. +8 ).
__device__ __forceinline__ v16h frag_rowmajor(const _Float16* p0, int ld,
                                              int rowBase, int kBase, int lane) {
  const _Float16* p = p0 + (size_t)(rowBase + (lane & 15)) * ld
                         + kBase + ((lane >> 4) << 3);
  v8h lo = *(const v8h*)p;
  v8h hi = *(const v8h*)(p + 16);
  v16h r;
#pragma unroll
  for (int i = 0; i < 8; ++i) { r[i] = lo[i]; r[8 + i] = hi[i]; }
  return r;
}

// ===========================================================================
// f32 -> f16 elementwise convert (Q,K,V inputs and weight matrices)
// ===========================================================================
__global__ void __launch_bounds__(256)
cvt_f16_kernel(const float* __restrict__ src, _Float16* __restrict__ dst,
               int n) {
  const size_t base = ((size_t)blockIdx.x * 256 + threadIdx.x) * 8;
  if (base >= (size_t)n) return;
  v4f a = *(const v4f*)(src + base);
  v4f b = *(const v4f*)(src + base + 4);
  v8h h;
#pragma unroll
  for (int i = 0; i < 4; ++i) {
    h[i]     = (_Float16)a[i];
    h[4 + i] = (_Float16)b[i];
  }
  *(v8h*)(dst + base) = h;
}

// ===========================================================================
// GEMM: out[M,1024] = A[M,1024] @ W[1024,1024]^T + bias, A/W in f16.
// 256 thr = 8 waves (2Mx4N); block tile 128x128, wave tile 64x32
// (8 WMMA / K-step). Async global->LDS staging, double-buffered.
// ===========================================================================
template <typename TO>
__global__ void __launch_bounds__(256)
gemm_f16_kernel(const _Float16* __restrict__ A, const _Float16* __restrict__ W,
                const float* __restrict__ bias, TO* __restrict__ out) {
  __shared__ __align__(16) _Float16 As[2][128 * 32];
  __shared__ __align__(16) _Float16 Bs[2][128 * 32];

  const int tid  = threadIdx.x;
  const int lane = tid & 31;
  const int wave = tid >> 5;
  const int wm   = wave & 1;
  const int wn   = wave >> 1;
  const int m0   = blockIdx.x * 128;
  const int n0   = blockIdx.y * 128;

  const int r0 = (tid + 0)   >> 2, s0 = ((tid + 0)   & 3) << 3;
  const int r1 = (tid + 256) >> 2, s1 = ((tid + 256) & 3) << 3;

  auto issue = [&](int kk, _Float16* dA, _Float16* dB) {
    async_b128(A + (size_t)(m0 + r0) * DMODEL + kk + s0, dA + r0 * 32 + s0);
    async_b128(A + (size_t)(m0 + r1) * DMODEL + kk + s1, dA + r1 * 32 + s1);
    async_b128(W + (size_t)(n0 + r0) * DMODEL + kk + s0, dB + r0 * 32 + s0);
    async_b128(W + (size_t)(n0 + r1) * DMODEL + kk + s1, dB + r1 * 32 + s1);
  };

  v8f acc[4][2] = {};

  int buf = 0;
  issue(0, As[0], Bs[0]);

  for (int k0 = 0; k0 < DMODEL; k0 += 32) {
    async_wait0();
    __syncthreads();

    if (k0 + 32 < DMODEL) issue(k0 + 32, As[buf ^ 1], Bs[buf ^ 1]);

    v16h bf0 = frag_rowmajor(Bs[buf], 32, wn * 32, 0, lane);
    v16h bf1 = frag_rowmajor(Bs[buf], 32, wn * 32 + 16, 0, lane);
#pragma unroll
    for (int mi = 0; mi < 4; ++mi) {
      v16h af = frag_rowmajor(As[buf], 32, wm * 64 + mi * 16, 0, lane);
      acc[mi][0] = wmma16(af, bf0, acc[mi][0]);
      acc[mi][1] = wmma16(af, bf1, acc[mi][1]);
    }

    __syncthreads();
    buf ^= 1;
  }

  const int col = lane & 15;
  const int rb  = (lane >> 4) << 3;
#pragma unroll
  for (int ni = 0; ni < 2; ++ni) {
    const int n = n0 + wn * 32 + ni * 16 + col;
    const float bv = bias[n];
#pragma unroll
    for (int mi = 0; mi < 4; ++mi) {
#pragma unroll
      for (int j = 0; j < 8; ++j) {
        out[(size_t)(m0 + wm * 64 + mi * 16 + rb + j) * DMODEL + n] =
            (TO)(acc[mi][ni][j] + bv);
      }
    }
  }
}

// ===========================================================================
// scores: writes raw scaled scores to attn AND per-(row, k-tile) partial
// softmax stats: pmax = max_c s, psum = sum_c exp(s - pmax).
// Grid: (S/64, S/64, B*H).
// ===========================================================================
__global__ void __launch_bounds__(256)
scores_kernel(const _Float16* __restrict__ qp, const _Float16* __restrict__ kp,
              float* __restrict__ attn,
              float* __restrict__ pmax, float* __restrict__ psum,
              float scale) {
  __shared__ float Ssm[64 * 68];   // padded score tile for row scans

  const int tid  = threadIdx.x;
  const int lane = tid & 31;
  const int wave = tid >> 5;
  const int wm   = wave & 1;
  const int wn   = wave >> 1;
  const int q0   = blockIdx.x * 64;
  const int kc0  = blockIdx.y * 64;
  const int bh   = blockIdx.z;
  const int b    = bh >> 4;
  const int h    = bh & 15;

  const _Float16* qb = qp + ((size_t)b * S_LEN) * DMODEL + h * DK;
  const _Float16* kb = kp + ((size_t)b * S_LEN) * DMODEL + h * DK;

  v8f acc0 = {};
  v8f acc1 = {};
#pragma unroll
  for (int d0 = 0; d0 < DK; d0 += 32) {
    v16h bf = frag_rowmajor(kb, DMODEL, kc0 + wn * 16, d0, lane);
    v16h a0 = frag_rowmajor(qb, DMODEL, q0 + wm * 32, d0, lane);
    v16h a1 = frag_rowmajor(qb, DMODEL, q0 + wm * 32 + 16, d0, lane);
    acc0 = wmma16(a0, bf, acc0);
    acc1 = wmma16(a1, bf, acc1);
  }

  float* orow = attn + (size_t)bh * S_LEN * S_LEN;
  const int col = lane & 15;
  const int rb  = (lane >> 4) << 3;
  const int n   = kc0 + wn * 16 + col;
#pragma unroll
  for (int j = 0; j < 8; ++j) {
    const float s0v = acc0[j] * scale;
    const float s1v = acc1[j] * scale;
    orow[(size_t)(q0 + wm * 32 + rb + j) * S_LEN + n]      = s0v;
    orow[(size_t)(q0 + wm * 32 + 16 + rb + j) * S_LEN + n] = s1v;
    Ssm[(wm * 32 + rb + j) * 68 + wn * 16 + col]           = s0v;
    Ssm[(wm * 32 + 16 + rb + j) * 68 + wn * 16 + col]      = s1v;
  }
  __syncthreads();

  // One thread per row: partial max + partial sum over this 64-col tile.
  if (tid < 64) {
    const float* r = Ssm + tid * 68;
    float m = r[0];
#pragma unroll 8
    for (int c = 1; c < 64; ++c) m = fmaxf(m, r[c]);
    float s = 0.f;
#pragma unroll 8
    for (int c = 0; c < 64; ++c) s += __expf(r[c] - m);
    const size_t pi = ((size_t)bh * S_LEN + q0 + tid) * NKT + blockIdx.y;
    pmax[pi] = m;
    psum[pi] = s;
  }
}

// ===========================================================================
// Combine partial softmax stats: rowM = max_j pmax_j,
// rowInvL = 1 / sum_j psum_j * exp(pmax_j - rowM).  One thread per row.
// ===========================================================================
__global__ void __launch_bounds__(256)
combine_kernel(const float* __restrict__ pmax, const float* __restrict__ psum,
               float* __restrict__ rowM, float* __restrict__ rowInvL) {
  const size_t row = (size_t)blockIdx.x * 256 + threadIdx.x;
  const float* pm = pmax + row * NKT;
  const float* ps = psum + row * NKT;
  float m = pm[0];
#pragma unroll
  for (int j = 1; j < NKT; ++j) m = fmaxf(m, pm[j]);
  float l = 0.f;
#pragma unroll
  for (int j = 0; j < NKT; ++j) l += ps[j] * __expf(pm[j] - m);
  rowM[row]    = m;
  rowInvL[row] = 1.0f / l;
}

// ===========================================================================
// Fused normalize + ctx GEMM:
//   p = exp(s - rowM) * rowInvL  (written back to attn in place, f32)
//   ctx[b,q,h*64+d] = sum_k p * v[b,h,k,d]   (p rounded to f16 for WMMA)
// Grid: (S/64, B*H). Block tile 64(q) x 64(d). V tile via async copy.
// ===========================================================================
__global__ void __launch_bounds__(256)
context_kernel(float* __restrict__ attn, const _Float16* __restrict__ vp,
               const float* __restrict__ rowM, const float* __restrict__ rowInvL,
               _Float16* __restrict__ ctx) {
  __shared__ __align__(16) _Float16 As[64 * 32];  // [q][k] probabilities f16
  __shared__ __align__(16) _Float16 Vs[32 * 64];  // [k][d]

  const int tid  = threadIdx.x;
  const int lane = tid & 31;
  const int wave = tid >> 5;
  const int wm   = wave & 1;
  const int wn   = wave >> 1;
  const int q0   = blockIdx.x * 64;
  const int bh   = blockIdx.y;
  const int b    = bh >> 4;
  const int h    = bh & 15;

  float*          ab = attn + (size_t)bh * S_LEN * S_LEN;
  const _Float16* vb = vp + ((size_t)b * S_LEN) * DMODEL + h * DK;

  const int ra = (tid * 8) >> 5, ca = (tid * 8) & 31;  // attn stage row/col
  const int rv = tid >> 3, cv = (tid & 7) << 3;        // v stage: 32x64 b128

  // Per-staging-thread softmax constants (row is fixed per thread).
  const float m  = rowM[(size_t)bh * S_LEN + q0 + ra];
  const float il = rowInvL[(size_t)bh * S_LEN + q0 + ra];

  v8f acc0 = {};
  v8f acc1 = {};

  for (int k0 = 0; k0 < S_LEN; k0 += 32) {
    async_b128(vb + (size_t)(k0 + rv) * DMODEL + cv, Vs + rv * 64 + cv);

    // Stage 8 raw scores -> probabilities; write back f32, stage f16.
    {
      float* src = ab + (size_t)(q0 + ra) * S_LEN + k0 + ca;
      v4f a = *(const v4f*)src;
      v4f c = *(const v4f*)(src + 4);
      v4f pa, pc;
      v8h hh;
#pragma unroll
      for (int i = 0; i < 4; ++i) {
        pa[i] = __expf(a[i] - m) * il;
        pc[i] = __expf(c[i] - m) * il;
      }
#pragma unroll
      for (int i = 0; i < 4; ++i) {
        hh[i]     = (_Float16)pa[i];
        hh[4 + i] = (_Float16)pc[i];
      }
      *(v4f*)src       = pa;   // attn output (normalized, in place)
      *(v4f*)(src + 4) = pc;
      *(v8h*)(As + ra * 32 + ca) = hh;
    }
    async_wait0();
    __syncthreads();

    // B fragment (32x16): column n = wn*16 + lane%16, K gather over rows.
    v16h bf;
    {
      const int nn  = wn * 16 + (lane & 15);
      const int kb2 = (lane >> 4) << 3;
#pragma unroll
      for (int i = 0; i < 8; ++i) {
        bf[i]     = Vs[(kb2 + i) * 64 + nn];
        bf[8 + i] = Vs[(kb2 + 16 + i) * 64 + nn];
      }
    }
    v16h a0 = frag_rowmajor(As, 32, wm * 32, 0, lane);
    v16h a1 = frag_rowmajor(As, 32, wm * 32 + 16, 0, lane);
    acc0 = wmma16(a0, bf, acc0);
    acc1 = wmma16(a1, bf, acc1);
    __syncthreads();
  }

  const int col = lane & 15;
  const int rb  = (lane >> 4) << 3;
  const int d   = h * DK + wn * 16 + col;
#pragma unroll
  for (int j = 0; j < 8; ++j) {
    ctx[((size_t)b * S_LEN + q0 + wm * 32 + rb + j) * DMODEL + d] =
        (_Float16)acc0[j];
    ctx[((size_t)b * S_LEN + q0 + wm * 32 + 16 + rb + j) * DMODEL + d] =
        (_Float16)acc1[j];
  }
}

// ===========================================================================
extern "C" void kernel_launch(void* const* d_in, const int* in_sizes, int n_in,
                              void* d_out, int out_size, void* d_ws,
                              size_t ws_size, hipStream_t stream) {
  (void)in_sizes; (void)n_in; (void)out_size; (void)ws_size;

  const float* Q  = (const float*)d_in[0];
  const float* K  = (const float*)d_in[1];
  const float* V  = (const float*)d_in[2];
  const float* Wq = (const float*)d_in[3];
  const float* bq = (const float*)d_in[4];
  const float* Wk = (const float*)d_in[5];
  const float* bk = (const float*)d_in[6];
  const float* Wv = (const float*)d_in[7];
  const float* bv = (const float*)d_in[8];
  const float* Wo = (const float*)d_in[9];
  const float* bo = (const float*)d_in[10];

  float* out  = (float*)d_out;                 // [4096,1024]
  float* attn = out + (size_t)MROWS * DMODEL;  // [4,16,1024,1024]

  const size_t NMD  = (size_t)MROWS * DMODEL;   // 4M elems
  const size_t NW   = (size_t)DMODEL * DMODEL;  // 1M elems
  const size_t ROWS = (size_t)BATCH * NHEAD * S_LEN;  // 65536

  _Float16* Qh  = (_Float16*)d_ws;
  _Float16* Kh  = Qh  + NMD;
  _Float16* Vh  = Kh  + NMD;
  _Float16* Wqh = Vh  + NMD;
  _Float16* Wkh = Wqh + NW;
  _Float16* Wvh = Wkh + NW;
  _Float16* Woh = Wvh + NW;
  _Float16* qp  = Woh + NW;
  _Float16* kp  = qp  + NMD;
  _Float16* vp  = kp  + NMD;
  _Float16* ctx = vp  + NMD;                   // 32M halves = 64 MB
  float* pmax    = (float*)(ctx + NMD);        // [ROWS][16]
  float* psum    = pmax + ROWS * NKT;          // [ROWS][16]
  float* rowM    = psum + ROWS * NKT;          // [ROWS]
  float* rowInvL = rowM + ROWS;                // [ROWS]  (~72.5 MB total)

  const dim3 blk(256);

  cvt_f16_kernel<<<dim3((unsigned)(NMD / 2048)), blk, 0, stream>>>(Q, Qh, (int)NMD);
  cvt_f16_kernel<<<dim3((unsigned)(NMD / 2048)), blk, 0, stream>>>(K, Kh, (int)NMD);
  cvt_f16_kernel<<<dim3((unsigned)(NMD / 2048)), blk, 0, stream>>>(V, Vh, (int)NMD);
  cvt_f16_kernel<<<dim3((unsigned)(NW / 2048)),  blk, 0, stream>>>(Wq, Wqh, (int)NW);
  cvt_f16_kernel<<<dim3((unsigned)(NW / 2048)),  blk, 0, stream>>>(Wk, Wkh, (int)NW);
  cvt_f16_kernel<<<dim3((unsigned)(NW / 2048)),  blk, 0, stream>>>(Wv, Wvh, (int)NW);
  cvt_f16_kernel<<<dim3((unsigned)(NW / 2048)),  blk, 0, stream>>>(Wo, Woh, (int)NW);

  const dim3 gProj(MROWS / 128, DMODEL / 128);  // 32 x 8
  gemm_f16_kernel<_Float16><<<gProj, blk, 0, stream>>>(Qh, Wqh, bq, qp);
  gemm_f16_kernel<_Float16><<<gProj, blk, 0, stream>>>(Kh, Wkh, bk, kp);
  gemm_f16_kernel<_Float16><<<gProj, blk, 0, stream>>>(Vh, Wvh, bv, vp);

  scores_kernel<<<dim3(S_LEN / 64, S_LEN / 64, BATCH * NHEAD), blk, 0, stream>>>(
      qp, kp, attn, pmax, psum, 0.125f /* 1/sqrt(64) */);

  combine_kernel<<<dim3((unsigned)(ROWS / 256)), blk, 0, stream>>>(
      pmax, psum, rowM, rowInvL);

  context_kernel<<<dim3(S_LEN / 64, BATCH * NHEAD), blk, 0, stream>>>(
      attn, vp, rowM, rowInvL, ctx);

  gemm_f16_kernel<float><<<gProj, blk, 0, stream>>>(ctx, Woh, bo, out);
}
